// Discrim_net_69157563400289
// MI455X (gfx1250) — matrix operand order repaired and checked by reading, not compiled, not run on hardware
//
#include <hip/hip_runtime.h>
#include <stdint.h>

// ---------------------------------------------------------------------------
// Types / helpers
// ---------------------------------------------------------------------------
typedef __attribute__((ext_vector_type(16))) __bf16 v16bf;
typedef __attribute__((ext_vector_type(8)))  float  v8f;

union AFrag { v16bf v; uint4    q[2]; };
union BFrag { v16bf v; uint32_t u[8]; };

static __device__ __forceinline__ __bf16 f2bf(float f) {
    union { float f; uint32_t u; } a; a.f = f;
    uint32_t r = a.u + 0x7FFFu + ((a.u >> 16) & 1u);   // round-to-nearest-even
    union { unsigned short s; __bf16 b; } o; o.s = (unsigned short)(r >> 16);
    return o.b;
}
static __device__ __forceinline__ float bf2f(__bf16 b) {
    union { unsigned short s; __bf16 b; } i; i.b = b;
    union { uint32_t u; float f; } o; o.u = ((uint32_t)i.s) << 16;
    return o.f;
}

#define B_BATCH 128
#define FQ      318
#define CH0     109
#define NNODES  1024
#define MSLOTS  1025

// ---------------------------------------------------------------------------
// init: zero stats + zero the null-child (slot 0) columns of t1/t2/t3
// ---------------------------------------------------------------------------
__global__ void init_kernel(float* stats, __bf16* t1, __bf16* t2, __bf16* t3) {
    int i = blockIdx.x * blockDim.x + threadIdx.x;
    __bf16 z = f2bf(0.0f);
    if (i < 768) { stats[i] = 0.0f; return; }
    int j = i - 768;
    if (j < B_BATCH * 512) { t1[(size_t)j * MSLOTS] = z; return; }
    j -= B_BATCH * 512;
    if (j < B_BATCH * 256) { t2[(size_t)j * MSLOTS] = z; return; }
    j -= B_BATCH * 256;
    if (j < B_BATCH * 128) { t3[(size_t)j * MSLOTS] = z; return; }
}

// ---------------------------------------------------------------------------
// Weight packing: fp32 (O, Cin, 3) -> bf16 (O, KPAD), K = c*3+kk, zero padded
// ---------------------------------------------------------------------------
__global__ void pack_conv_kernel(const float* __restrict__ w, __bf16* __restrict__ dst,
                                 int O, int Klog, int KPAD) {
    int total = O * KPAD;
    for (int i = blockIdx.x * blockDim.x + threadIdx.x; i < total;
         i += gridDim.x * blockDim.x) {
        int o = i / KPAD, k = i - o * KPAD;
        float v = (k < Klog) ? w[o * Klog + k] : 0.0f;
        dst[i] = f2bf(v);
    }
}

// W_enc (109,318) -> bf16 (112,320) zero padded
__global__ void pack_enc_kernel(const float* __restrict__ w, __bf16* __restrict__ dst) {
    int total = 112 * 320;
    for (int i = blockIdx.x * blockDim.x + threadIdx.x; i < total;
         i += gridDim.x * blockDim.x) {
        int c = i / 320, f = i - c * 320;
        float v = (c < CH0 && f < FQ) ? w[c * FQ + f] : 0.0f;
        dst[i] = f2bf(v);
    }
}

// ---------------------------------------------------------------------------
// Encoder: t0[b,c,m] = sum_f Wenc[c,f] * x[b,f,m] + b_enc[c]; slot0 = 0
// grid: (65 column tiles, B), block: 256 (8 waves)
// ---------------------------------------------------------------------------
__global__ void encoder_kernel(const float* __restrict__ x,
                               const __bf16* __restrict__ wEnc,
                               const float* __restrict__ bEnc,
                               __bf16* __restrict__ t0) {
    constexpr int KPAD = 320;
    __shared__ __align__(16) __bf16 g[KPAD * 16];

    const int b   = blockIdx.y;
    const int mt  = blockIdx.x;
    const int tid = threadIdx.x;

    // Stage B panel (K x 16 columns of x, converted to bf16) into LDS,
    // pre-packed so K-pairs are dword-contiguous per column.
    for (int e = tid; e < KPAD * 16; e += 256) {
        int k = e >> 4, j = e & 15;
        int m = mt * 16 + j;
        float v = 0.0f;
        if (k < FQ && m < MSLOTS) v = x[((size_t)b * FQ + k) * MSLOTS + m];
        g[((k >> 1) << 5) + (j << 1) + (k & 1)] = f2bf(v);
    }
    __syncthreads();

    const int wv   = tid >> 5;
    const int lane = tid & 31;
    const int j    = lane & 15;
    const int kbA  = (lane < 16) ? 0 : 16;
    const int kbB  = (lane < 16) ? 0 : 16;
    const int hi8  = (lane < 16) ? 0 : 8;
    const uint32_t* lds32 = (const uint32_t*)g;

    for (int ot = wv; ot < 7; ot += 8) {
        const int row = ot * 16 + j;          // padded weight row (< 112)
        v8f acc; for (int r = 0; r < 8; ++r) acc[r] = 0.0f;
        for (int ks = 0; ks < KPAD; ks += 32) {
            AFrag a;
            const uint4* pa = (const uint4*)(wEnc + (size_t)row * KPAD + ks + kbA);
            a.q[0] = pa[0];            // K = kbA .. kbA+7
            a.q[1] = pa[1];            // K = kbA+8 .. kbA+15
            BFrag bb;
            int base = ((ks + kbB) >> 1) * 16 + j;
            #pragma unroll
            for (int v = 0; v < 8; ++v) bb.u[v] = lds32[base + v * 16];
            acc = __builtin_amdgcn_wmma_f32_16x16x32_bf16(
                false, a.v, false, bb.v, (short)0, acc, false, false);
        }
        const int m = mt * 16 + j;
        if (m < MSLOTS) {
            #pragma unroll
            for (int r = 0; r < 8; ++r) {
                int oc = ot * 16 + r + hi8;
                if (oc < CH0) {
                    float val = (m == 0) ? 0.0f : (acc[r] + bEnc[oc]);
                    t0[((size_t)b * CH0 + oc) * MSLOTS + m] = f2bf(val);
                }
            }
        }
    }
}

// ---------------------------------------------------------------------------
// Gather + tree conv as a bf16 WMMA GEMM.
// out[b,o,n+1] = sum_{c,kk} W[o,c,kk] * tin[b,c, idx[b, 3n+kk]] + bias[o]
// grid: (64 node tiles, B), block 256 (8 waves). Accumulates layer norm stats.
// ---------------------------------------------------------------------------
template <int CIN, int COUT, int KPAD>
__global__ void conv_btc_kernel(const __bf16* __restrict__ tin,
                                const int*    __restrict__ idx,
                                const __bf16* __restrict__ wK,
                                const float*  __restrict__ bias,
                                __bf16*       __restrict__ tout,
                                float*        __restrict__ stats) {
    __shared__ __align__(16) __bf16 g[KPAD * 16];
    __shared__ int sidx[48];

    const int b   = blockIdx.y;
    const int nt  = blockIdx.x;
    const int tid = threadIdx.x;
    constexpr int KLOG = 3 * CIN;

    if (tid < 48) sidx[tid] = idx[(size_t)b * (3 * NNODES) + nt * 48 + tid];
    __syncthreads();

    // Gather K x 16 panel from tin through the (node,left,right) triples.
    for (int e = tid; e < KPAD * 16; e += 256) {
        int k = e >> 4, j = e & 15;
        __bf16 v = f2bf(0.0f);
        if (k < KLOG) {
            int c  = k / 3;
            int kk = k - 3 * c;
            int iv = sidx[j * 3 + kk];                      // 0..1024, 0 = null
            v = tin[((size_t)b * CIN + c) * MSLOTS + iv];
        }
        g[((k >> 1) << 5) + (j << 1) + (k & 1)] = v;
    }
    __syncthreads();

    const int wv   = tid >> 5;
    const int lane = tid & 31;
    const int j    = lane & 15;
    const int kbA  = (lane < 16) ? 0 : 16;
    const int kbB  = (lane < 16) ? 0 : 16;
    const int hi8  = (lane < 16) ? 0 : 8;
    const uint32_t* lds32 = (const uint32_t*)g;

    for (int ot = wv; ot < COUT / 16; ot += 8) {
        const int row = ot * 16 + j;
        v8f acc; for (int r = 0; r < 8; ++r) acc[r] = 0.0f;
        for (int ks = 0; ks < KPAD; ks += 32) {
            AFrag a;
            const uint4* pa = (const uint4*)(wK + (size_t)row * KPAD + ks + kbA);
            a.q[0] = pa[0];
            a.q[1] = pa[1];
            BFrag bb;
            int base = ((ks + kbB) >> 1) * 16 + j;
            #pragma unroll
            for (int v = 0; v < 8; ++v) bb.u[v] = lds32[base + v * 16];
            acc = __builtin_amdgcn_wmma_f32_16x16x32_bf16(
                false, a.v, false, bb.v, (short)0, acc, false, false);
        }
        const int m = nt * 16 + j + 1;                     // slot 0 reserved
        float s1 = 0.0f, s2 = 0.0f;
        #pragma unroll
        for (int r = 0; r < 8; ++r) {
            int oc = ot * 16 + r + hi8;
            float f = acc[r] + bias[oc];
            s1 += f; s2 += f * f;
            tout[((size_t)b * COUT + oc) * MSLOTS + m] = f2bf(f);
        }
        // wave reduction for layer-norm statistics
        #pragma unroll
        for (int off = 16; off > 0; off >>= 1) {
            s1 += __shfl_xor(s1, off, 32);
            s2 += __shfl_xor(s2, off, 32);
        }
        if (lane == 0) {
            atomicAdd(&stats[b * 2 + 0], s1);
            atomicAdd(&stats[b * 2 + 1], s2);
        }
    }
}

// ---------------------------------------------------------------------------
// TreeLayerNorm (unbiased std) + relu, in place over bf16 activations.
// ---------------------------------------------------------------------------
__global__ void normalize_kernel(__bf16* __restrict__ t,
                                 const float* __restrict__ stats,
                                 int C, float nElems) {
    size_t total = (size_t)B_BATCH * C * MSLOTS;
    size_t stride = (size_t)gridDim.x * blockDim.x;
    for (size_t i = (size_t)blockIdx.x * blockDim.x + threadIdx.x; i < total; i += stride) {
        int b = (int)(i / ((size_t)C * MSLOTS));
        float s1 = stats[b * 2 + 0];
        float s2 = stats[b * 2 + 1];
        float mean = s1 / nElems;
        float var  = (s2 - s1 * s1 / nElems) / (nElems - 1.0f);
        var = var > 0.0f ? var : 0.0f;
        float inv = 1.0f / (sqrtf(var) + 1e-5f);
        float v = (bf2f(t[i]) - mean) * inv;
        t[i] = f2bf(v > 0.0f ? v : 0.0f);
    }
}

// ---------------------------------------------------------------------------
// Max pool over nodes + 4-layer MLP head. One block per batch row.
// ---------------------------------------------------------------------------
__global__ void head_kernel(const __bf16* __restrict__ t3,
                            const float* __restrict__ Wh1, const float* __restrict__ bh1,
                            const float* __restrict__ Wh2, const float* __restrict__ bh2,
                            const float* __restrict__ Wh3, const float* __restrict__ bh3,
                            const float* __restrict__ Wh4, const float* __restrict__ bh4,
                            float* __restrict__ out) {
    __shared__ float pool[128], h1[128], h2[64], h3[32];
    const int b = blockIdx.x, tid = threadIdx.x;

    float mx = -1e30f;
    size_t base = ((size_t)b * 128 + tid) * MSLOTS;
    for (int m = 0; m < MSLOTS; ++m) mx = fmaxf(mx, bf2f(t3[base + m]));
    pool[tid] = mx;
    __syncthreads();

    { float s = bh1[tid];
      for (int f = 0; f < 128; ++f) s += Wh1[tid * 128 + f] * pool[f];
      h1[tid] = fmaxf(s, 0.0f); }
    __syncthreads();
    if (tid < 64) {
        float s = bh2[tid];
        for (int f = 0; f < 128; ++f) s += Wh2[tid * 128 + f] * h1[f];
        h2[tid] = fmaxf(s, 0.0f);
    }
    __syncthreads();
    if (tid < 32) {
        float s = bh3[tid];
        for (int f = 0; f < 64; ++f) s += Wh3[tid * 64 + f] * h2[f];
        h3[tid] = fmaxf(s, 0.0f);
    }
    __syncthreads();
    if (tid == 0) {
        float s = bh4[0];
        for (int f = 0; f < 32; ++f) s += Wh4[f] * h3[f];
        out[b] = s;
    }
}

// ---------------------------------------------------------------------------
// Launch
// ---------------------------------------------------------------------------
static inline size_t alignUp(size_t x) { return (x + 255) & ~(size_t)255; }

extern "C" void kernel_launch(void* const* d_in, const int* in_sizes, int n_in,
                              void* d_out, int out_size, void* d_ws, size_t ws_size,
                              hipStream_t stream) {
    const float* x     = (const float*)d_in[0];
    const int*   idx   = (const int*)  d_in[1];
    const float* Wenc  = (const float*)d_in[2];
    const float* benc  = (const float*)d_in[3];
    const float* W1    = (const float*)d_in[4];
    const float* b1    = (const float*)d_in[5];
    const float* W2    = (const float*)d_in[6];
    const float* b2    = (const float*)d_in[7];
    const float* W3    = (const float*)d_in[8];
    const float* b3    = (const float*)d_in[9];
    const float* Wh1   = (const float*)d_in[10];
    const float* bh1   = (const float*)d_in[11];
    const float* Wh2   = (const float*)d_in[12];
    const float* bh2   = (const float*)d_in[13];
    const float* Wh3   = (const float*)d_in[14];
    const float* bh3   = (const float*)d_in[15];
    const float* Wh4   = (const float*)d_in[16];
    const float* bh4   = (const float*)d_in[17];
    float* out = (float*)d_out;

    // Workspace layout (bf16 unless noted)
    char* ws = (char*)d_ws;
    size_t off = 0;
    __bf16* wEncP = (__bf16*)(ws + off); off = alignUp(off + (size_t)112 * 320 * 2);
    __bf16* w1P   = (__bf16*)(ws + off); off = alignUp(off + (size_t)512 * 352 * 2);
    __bf16* w2P   = (__bf16*)(ws + off); off = alignUp(off + (size_t)256 * 1536 * 2);
    __bf16* w3P   = (__bf16*)(ws + off); off = alignUp(off + (size_t)128 * 768 * 2);
    __bf16* t0    = (__bf16*)(ws + off); off = alignUp(off + (size_t)B_BATCH * CH0 * MSLOTS * 2);
    __bf16* t1    = (__bf16*)(ws + off); off = alignUp(off + (size_t)B_BATCH * 512 * MSLOTS * 2);
    __bf16* t2    = (__bf16*)(ws + off); off = alignUp(off + (size_t)B_BATCH * 256 * MSLOTS * 2);
    __bf16* t3    = (__bf16*)(ws + off); off = alignUp(off + (size_t)B_BATCH * 128 * MSLOTS * 2);
    float*  stats = (float*)(ws + off);  off = alignUp(off + (size_t)3 * 256 * 4);

    // 0) init stats + slot-0 columns
    {
        int total = 768 + B_BATCH * (512 + 256 + 128);
        init_kernel<<<(total + 255) / 256, 256, 0, stream>>>(stats, t1, t2, t3);
    }
    // 1) pack weights to bf16, K-flattened + padded
    pack_enc_kernel<<<140, 256, 0, stream>>>(Wenc, wEncP);
    pack_conv_kernel<<<512, 256, 0, stream>>>(W1, w1P, 512, 3 * 109, 352);
    pack_conv_kernel<<<512, 256, 0, stream>>>(W2, w2P, 256, 3 * 512, 1536);
    pack_conv_kernel<<<512, 256, 0, stream>>>(W3, w3P, 128, 3 * 256, 768);

    // 2) encoder
    encoder_kernel<<<dim3(65, B_BATCH), 256, 0, stream>>>(x, wEncP, benc, t0);

    // 3) conv1 + norm
    conv_btc_kernel<109, 512, 352><<<dim3(64, B_BATCH), 256, 0, stream>>>(
        t0, idx, w1P, b1, t1, stats + 0);
    normalize_kernel<<<2048, 256, 0, stream>>>(t1, stats + 0, 512, 512.0f * MSLOTS);

    // 4) conv2 + norm
    conv_btc_kernel<512, 256, 1536><<<dim3(64, B_BATCH), 256, 0, stream>>>(
        t1, idx, w2P, b2, t2, stats + 256);
    normalize_kernel<<<2048, 256, 0, stream>>>(t2, stats + 256, 256, 256.0f * MSLOTS);

    // 5) conv3 + norm
    conv_btc_kernel<256, 128, 768><<<dim3(64, B_BATCH), 256, 0, stream>>>(
        t2, idx, w3P, b3, t3, stats + 512);
    normalize_kernel<<<1024, 256, 0, stream>>>(t3, stats + 512, 128, 128.0f * MSLOTS);

    // 6) pool + MLP head
    head_kernel<<<B_BATCH, 128, 0, stream>>>(t3, Wh1, bh1, Wh2, bh2, Wh3, bh3,
                                             Wh4, bh4, out);
}
